// AFGSA_56212531970080
// MI455X (gfx1250) — compile-verified, matrix operand order checked by reading, not run on hardware
//
#include <hip/hip_runtime.h>
#include <hip/hip_bf16.h>

typedef __bf16 bf16_t;
typedef __attribute__((ext_vector_type(16))) __bf16 v16bf;
typedef __attribute__((ext_vector_type(8)))  __bf16 v8bf;
typedef __attribute__((ext_vector_type(8)))  float  v8f;

#define WMMA_BF16(a, b, c) \
  __builtin_amdgcn_wmma_f32_16x16x32_bf16(false, (a), false, (b), (short)0, (c), false, false)

#if __has_builtin(__builtin_amdgcn_global_load_async_to_lds_b128)
#define HAS_ASYNC_LDS 1
#else
#define HAS_ASYNC_LDS 0
#endif

#if HAS_ASYNC_LDS
// Builtin signature (from hipcc diagnostic): params are pointers to
// int __attribute__((vector_size(16))) in AS1 (global) / AS3 (LDS).
typedef int v4i_vs __attribute__((vector_size(16)));
typedef __attribute__((address_space(1))) v4i_vs* gptr4_t;
typedef __attribute__((address_space(3))) v4i_vs* lptr4_t;
#define AS1P(p) ((__attribute__((address_space(1))) void*)((void*)(p)))
#define AS3P(p) ((__attribute__((address_space(3))) void*)((void*)(p)))
__device__ __forceinline__ void async_copy16(const bf16_t* g, bf16_t* l) {
  __builtin_amdgcn_global_load_async_to_lds_b128((gptr4_t)AS1P(g), (lptr4_t)AS3P(l), 0, 0);
}
__device__ __forceinline__ void async_wait0() {
#if __has_builtin(__builtin_amdgcn_s_wait_asynccnt)
  __builtin_amdgcn_s_wait_asynccnt(0);
#else
  asm volatile("s_wait_asynccnt 0x0" ::: "memory");
#endif
}
#endif

// Load a 16x32 bf16 WMMA A-fragment (or B^T-fragment) from a row-major
// [rows][ld] bf16 array, tile origin = base. Per ISA 7.12.2:
//   row  m = lane & 15
//   Kbase  = (lane >= 16) ? 8 : 0 ; elements = {Kb..Kb+7} U {Kb+16..Kb+23}
__device__ __forceinline__ v16bf load_frag(const bf16_t* base, int ld) {
  const int lane = threadIdx.x & 31;
  const int m  = lane & 15;
  const int kb = (lane >> 4) << 3;
  const bf16_t* p = base + m * ld + kb;
  union { v16bf v; v8bf h[2]; } u;
  u.h[0] = *(const v8bf*)(p);
  u.h[1] = *(const v8bf*)(p + 16);
  return u.v;
}

// ---------------------------------------------------------------------------
// Pack [noisy;aux] (NCHW fp32) -> xin (pixel-major bf16, [b*HW + yx][512]).
// Coalesced reads; scattered bf16 writes merge in the 192MB L2 (xin = 67MB).
// ---------------------------------------------------------------------------
__global__ __launch_bounds__(256) void pack_xin_kernel(
    const float* __restrict__ noisy, const float* __restrict__ aux,
    bf16_t* __restrict__ xin) {
  int idx = blockIdx.x * 256 + threadIdx.x;     // 512 * 65536 total
  int c = idx >> 16;                            // 0..511
  int p = idx & 65535;                          // b*16384 + y*128 + x
  int b = p >> 14, yx = p & 16383;
  float v = (c < 256) ? noisy[((size_t)(b * 256 + c) << 14) + yx]
                      : aux  [((size_t)(b * 256 + (c - 256)) << 14) + yx];
  xin[(size_t)p * 512 + c] = (bf16_t)v;
}

__global__ __launch_bounds__(256) void cvt_bf16_kernel(
    const float* __restrict__ s, bf16_t* __restrict__ d, int n) {
  int i = blockIdx.x * 256 + threadIdx.x;
  if (i < n) d[i] = (bf16_t)s[i];
}

// ---------------------------------------------------------------------------
// Generic pixel-GEMM: D[p][o] = act(scale * sum_c A[p][c]*W[o][c] + bias[o])
// WG = 128 pixels x 256 outputs, 8 waves (wave w owns M-tile w = rows it
// itself stages into LDS, so the async path needs no block barrier at all).
// K streamed through a double-buffered LDS tile via async global->LDS DMA.
// ---------------------------------------------------------------------------
__global__ __launch_bounds__(256) void gemm_bf16(
    const bf16_t* __restrict__ A, int ldA, int K,
    const bf16_t* __restrict__ W,
    const float* __restrict__ bias, int relu, float scale,
    bf16_t* __restrict__ D) {
  __shared__ bf16_t As[2][128 * 64];
  const int pix0 = blockIdx.x * 128;
  const int t = threadIdx.x;
  const int wave = t >> 5, lane = t & 31;
  const int n = lane & 15, mo = (lane >> 4) << 3;

  v8f acc[16];
#pragma unroll
  for (int i = 0; i < 16; ++i) acc[i] = {};

  const int row = t >> 1, half = t & 1;     // wave w stages rows 16w..16w+15
  const int nChunk = K >> 6;
  const bf16_t* arow = A + (size_t)(pix0 + row) * ldA + half * 32;

#if HAS_ASYNC_LDS
  {  // prologue: chunk 0 -> buffer 0 (4 x b128 per thread, ASYNCcnt-tracked)
    bf16_t* d = &As[0][row * 64 + half * 32];
#pragma unroll
    for (int j = 0; j < 4; ++j) async_copy16(arow + j * 8, d + j * 8);
  }
  for (int kc = 0; kc < nChunk; ++kc) {
    async_wait0();                           // own rows are all this wave reads
    if (kc + 1 < nChunk) {                   // overlap next copy with WMMAs
      const bf16_t* s = arow + (kc + 1) * 64;
      bf16_t* d = &As[(kc + 1) & 1][row * 64 + half * 32];
#pragma unroll
      for (int j = 0; j < 4; ++j) async_copy16(s + j * 8, d + j * 8);
    }
    const bf16_t* Abuf = &As[kc & 1][0];
#pragma unroll
    for (int ks = 0; ks < 2; ++ks) {
      v16bf a = load_frag(Abuf + wave * 16 * 64 + ks * 32, 64);
#pragma unroll
      for (int nt = 0; nt < 16; ++nt) {
        v16bf b = load_frag(W + (size_t)(nt * 16) * K + kc * 64 + ks * 32, K);
        acc[nt] = WMMA_BF16(a, b, acc[nt]);
      }
    }
  }
#else
  for (int kc = 0; kc < nChunk; ++kc) {
    __syncthreads();
    {
      const uint4* s = (const uint4*)(arow + kc * 64);
      uint4* d = (uint4*)(&As[0][row * 64 + half * 32]);
      d[0] = s[0]; d[1] = s[1]; d[2] = s[2]; d[3] = s[3];
    }
    __syncthreads();
#pragma unroll
    for (int ks = 0; ks < 2; ++ks) {
      v16bf a = load_frag(&As[0][wave * 16 * 64 + ks * 32], 64);
#pragma unroll
      for (int nt = 0; nt < 16; ++nt) {
        v16bf b = load_frag(W + (size_t)(nt * 16) * K + kc * 64 + ks * 32, K);
        acc[nt] = WMMA_BF16(a, b, acc[nt]);
      }
    }
  }
#endif

#pragma unroll
  for (int nt = 0; nt < 16; ++nt) {
    float bv = bias ? bias[nt * 16 + n] : 0.f;
#pragma unroll
    for (int r = 0; r < 8; ++r) {
      float v = acc[nt][r] * scale + bv;
      if (relu) v = v > 0.f ? v : 0.f;
      D[(size_t)(pix0 + wave * 16 + mo + r) * 256 + nt * 16 + n] = (bf16_t)v;
    }
  }
}

// ---------------------------------------------------------------------------
// Fused halo attention: one WG per (b, by, bx, head); 8 waves.
// win=14 -> 196 key positions padded to 224 (7 K-steps of 32).
// ---------------------------------------------------------------------------
__global__ __launch_bounds__(256) void halo_attn(
    const bf16_t* __restrict__ qg, const bf16_t* __restrict__ kg,
    const bf16_t* __restrict__ vg,
    const float* __restrict__ rel_h, const float* __restrict__ rel_w,
    float* __restrict__ out) {
  extern __shared__ char smem[];
  bf16_t* q_s   = (bf16_t*)smem;            // 64 x 64
  bf16_t* k_s   = q_s  + 64 * 64;           // 224 x 64  (pos-major)
  bf16_t* vT_s  = k_s  + 224 * 64;          // 64 x 224  (d-major = B^T layout)
  bf16_t* at_s  = vT_s + 64 * 224;          // 64 x 224  attn bf16
  float*  sim_s = (float*)(at_s + 64 * 224);// 64 x 224  logits / staging
  float*  red_s = sim_s + 64 * 224;         // 64 x 4    reductions

  const int bid = blockIdx.x;
  const int h  = bid & 3;
  const int bx = (bid >> 2) & 15;
  const int by = (bid >> 6) & 15;
  const int b  = bid >> 10;
  const int t = threadIdx.x;
  const int wave = t >> 5, lane = t & 31;
  const int n = lane & 15, mo = (lane >> 4) << 3;

  {  // q tile: 64 query positions x d=64; async so it overlaps the k/v fills
    int pos = t >> 2, ch = (t & 3) << 4;
    const bf16_t* s = qg +
        ((size_t)((b * 128 + by * 8 + (pos >> 3)) * 128 + bx * 8 + (pos & 7)) * 256
         + h * 64 + ch);
    bf16_t* d = q_s + pos * 64 + ch;
#if HAS_ASYNC_LDS
    async_copy16(s, d);
    async_copy16(s + 8, d + 8);
#else
    ((uint4*)d)[0] = ((const uint4*)s)[0];
    ((uint4*)d)[1] = ((const uint4*)s)[1];
#endif
  }
  // k window + relative positional embedding (zero-padded like the reference)
  for (int i = t; i < 224 * 64; i += 256) {
    int pos = i >> 6, dc = i & 63;
    bf16_t o = (bf16_t)0.f;
    if (pos < 196) {
      int wy = pos / 14, wx = pos % 14;
      int gy = by * 8 - 3 + wy, gx = bx * 8 - 3 + wx;
      float kv = 0.f;
      if ((unsigned)gy < 128u && (unsigned)gx < 128u)
        kv = (float)kg[((size_t)((b * 128 + gy) * 128 + gx)) * 256 + h * 64 + dc];
      float rel = (dc < 32) ? rel_h[wy * 32 + dc] : rel_w[wx * 32 + (dc - 32)];
      o = (bf16_t)(kv + rel);
    }
    k_s[i] = o;
  }
  // v window, transposed so AV B-fragments read contiguous K
  for (int i = t; i < 64 * 224; i += 256) {
    int dc = i / 224, pos = i % 224;
    bf16_t o = (bf16_t)0.f;
    if (pos < 196) {
      int wy = pos / 14, wx = pos % 14;
      int gy = by * 8 - 3 + wy, gx = bx * 8 - 3 + wx;
      if ((unsigned)gy < 128u && (unsigned)gx < 128u)
        o = vg[((size_t)((b * 128 + gy) * 128 + gx)) * 256 + h * 64 + dc];
    }
    vT_s[dc * 224 + pos] = o;
  }
#if HAS_ASYNC_LDS
  async_wait0();
#endif
  __syncthreads();

  // sim = q * k^T : 4 x 14 tiles over 8 waves, K=64
#pragma unroll
  for (int i = 0; i < 7; ++i) {
    int tile = wave * 7 + i;
    int mt = tile / 14, nt = tile % 14;
    v8f acc = {};
#pragma unroll
    for (int ks = 0; ks < 2; ++ks) {
      v16bf a  = load_frag(q_s + mt * 16 * 64 + ks * 32, 64);
      v16bf bb = load_frag(k_s + nt * 16 * 64 + ks * 32, 64);
      acc = WMMA_BF16(a, bb, acc);
    }
#pragma unroll
    for (int r = 0; r < 8; ++r)
      sim_s[(mt * 16 + mo + r) * 224 + nt * 16 + n] = acc[r];
  }
  __syncthreads();

  // softmax over the 196 real columns (4 threads per row, 49 cols each)
  {
    int row = t >> 2, q4 = t & 3;
    int j0 = q4 * 49;
    float mx = -1e30f;
    for (int j = 0; j < 49; ++j) {
      float v = sim_s[row * 224 + j0 + j];
      mx = v > mx ? v : mx;
    }
    red_s[row * 4 + q4] = mx;
    __syncthreads();
    mx = fmaxf(fmaxf(red_s[row * 4 + 0], red_s[row * 4 + 1]),
               fmaxf(red_s[row * 4 + 2], red_s[row * 4 + 3]));
    __syncthreads();
    float sum = 0.f;
    for (int j = 0; j < 49; ++j) {
      float e = __expf(sim_s[row * 224 + j0 + j] - mx);
      sim_s[row * 224 + j0 + j] = e;
      sum += e;
    }
    red_s[row * 4 + q4] = sum;
    __syncthreads();
    float inv = 1.f / (red_s[row * 4 + 0] + red_s[row * 4 + 1] +
                       red_s[row * 4 + 2] + red_s[row * 4 + 3]);
    for (int j = 0; j < 49; ++j)
      at_s[row * 224 + j0 + j] = (bf16_t)(sim_s[row * 224 + j0 + j] * inv);
    if (q4 == 3)
      for (int j = 196; j < 224; ++j) at_s[row * 224 + j] = (bf16_t)0.f;
  }
  __syncthreads();

  // out = attn * v : 4 x 4 tiles over 8 waves, K=224 (pad contributes 0)
#pragma unroll
  for (int i = 0; i < 2; ++i) {
    int tile = wave * 2 + i;
    int mt = tile >> 2, nt = tile & 3;
    v8f acc = {};
#pragma unroll
    for (int ks = 0; ks < 7; ++ks) {
      v16bf a  = load_frag(at_s + mt * 16 * 224 + ks * 32, 224);
      v16bf bb = load_frag(vT_s + nt * 16 * 224 + ks * 32, 224);
      acc = WMMA_BF16(a, bb, acc);
    }
#pragma unroll
    for (int r = 0; r < 8; ++r)
      sim_s[(mt * 16 + mo + r) * 64 + nt * 16 + n] = acc[r];  // stage [pos][dc]
  }
  __syncthreads();

  // coalesced NCHW writeout (8 contiguous gx per 8 threads)
  for (int i = t; i < 64 * 64; i += 256) {
    int dcl = i >> 6, pos = i & 63;
    int gy = by * 8 + (pos >> 3), gx = bx * 8 + (pos & 7);
    out[((size_t)(b * 256 + h * 64 + dcl) * 128 + gy) * 128 + gx] =
        sim_s[pos * 64 + dcl];
  }
}

// ---------------------------------------------------------------------------
extern "C" void kernel_launch(void* const* d_in, const int* in_sizes, int n_in,
                              void* d_out, int out_size, void* d_ws, size_t ws_size,
                              hipStream_t stream) {
  const float* noisy = (const float*)d_in[0];
  const float* aux   = (const float*)d_in[1];
  const float* w_map = (const float*)d_in[2];
  const float* b_map = (const float*)d_in[3];
  const float* w_q   = (const float*)d_in[4];
  const float* w_k   = (const float*)d_in[5];
  const float* w_v   = (const float*)d_in[6];
  const float* rel_h = (const float*)d_in[7];
  const float* rel_w = (const float*)d_in[8];
  float* out = (float*)d_out;

  // workspace layout (bytes); total ~202 MB
  char* ws = (char*)d_ws;
  bf16_t* xin   = (bf16_t*)(ws + 0);            //  67,108,864  [p][512]
  bf16_t* naux  = (bf16_t*)(ws + 67108864);     //  33,554,432  [p][256]
  bf16_t* qg    = (bf16_t*)(ws + 100663296);    //  33,554,432
  bf16_t* kg    = (bf16_t*)(ws + 134217728);    //  33,554,432
  bf16_t* vg    = (bf16_t*)(ws + 167772160);    //  33,554,432
  bf16_t* wmapb = (bf16_t*)(ws + 201326592);    //     262,144  [256][512]
  bf16_t* wqb   = (bf16_t*)(ws + 201588736);    //     131,072  [256][256]
  bf16_t* wkb   = (bf16_t*)(ws + 201719808);
  bf16_t* wvb   = (bf16_t*)(ws + 201850880);

  pack_xin_kernel<<<131072, 256, 0, stream>>>(noisy, aux, xin);
  cvt_bf16_kernel<<<512, 256, 0, stream>>>(w_map, wmapb, 256 * 512);
  cvt_bf16_kernel<<<256, 256, 0, stream>>>(w_q, wqb, 256 * 256);
  cvt_bf16_kernel<<<256, 256, 0, stream>>>(w_k, wkb, 256 * 256);
  cvt_bf16_kernel<<<256, 256, 0, stream>>>(w_v, wvb, 256 * 256);

  // n_aux = relu(W_map . xin + b)
  gemm_bf16<<<512, 256, 0, stream>>>(xin, 512, 512, wmapb, b_map, 1, 1.0f, naux);
  // q (scaled by d^-0.5 = 0.125), k from n_aux; v from noisy (first 256 of xin)
  gemm_bf16<<<512, 256, 0, stream>>>(naux, 256, 256, wqb, nullptr, 0, 0.125f, qg);
  gemm_bf16<<<512, 256, 0, stream>>>(naux, 256, 256, wkb, nullptr, 0, 1.0f, kg);
  gemm_bf16<<<512, 256, 0, stream>>>(xin, 512, 256, wvb, nullptr, 0, 1.0f, vg);

  size_t shmem = (size_t)(64 * 64 + 224 * 64 + 64 * 224 + 64 * 224) * sizeof(bf16_t)
               + (size_t)(64 * 224 + 64 * 4) * sizeof(float);   // 152,576 B
  halo_attn<<<4096, 256, shmem, stream>>>(qg, kg, vg, rel_h, rel_w, out);
}
